// GraphWaveNetBlock_55379308315168
// MI455X (gfx1250) — matrix operand order repaired and checked by reading, not compiled, not run on hardware
//
#include <hip/hip_runtime.h>
#include <hip/hip_bf16.h>
#include <math.h>

// ---------------------------------------------------------------------------
// GraphWaveNet block on MI455X (gfx1250, wave32, WMMA 16x16x32 bf16 + TDM)
// Fully fused: out is written exactly once; hop-1 results never leave LDS.
// ---------------------------------------------------------------------------
typedef __attribute__((ext_vector_type(16))) __bf16        v16bf;
typedef __attribute__((ext_vector_type(8)))  float         v8f;
typedef __attribute__((ext_vector_type(4)))  unsigned int  v4u;
typedef __attribute__((ext_vector_type(4)))  unsigned int  u32x4;
typedef __attribute__((ext_vector_type(8)))  int           i32x8;
typedef __attribute__((ext_vector_type(4)))  int           i32x4;

union Frag { v16bf v; v4u q[2]; __bf16 e[16]; };

#define B_  8
#define T_  256
#define N_  512
#define C_  64
#define ROWS ((size_t)B_ * T_ * N_)   // 1,048,576
#define ALPHA_ 0.05f

__device__ __host__ __forceinline__ int kmapA(int lane, int h) {
  return h + ((lane & 16) ? 16 : 0);
}
__device__ __host__ __forceinline__ int kmapB(int lane, int h) {
  return (h < 8 ? h : h + 8) + ((lane & 16) ? 8 : 0);
}

__device__ __forceinline__ v8f wmma_bf16(const Frag& a, const Frag& b, v8f c) {
  return __builtin_amdgcn_wmma_f32_16x16x32_bf16(false, a.v, false, b.v,
                                                 (short)0, c, false, false);
}

// CDNA5 transcendental tanh (TRANS32 op); trailing v_nop covers the
// trans-op result hazard per ISA 15.5 scheduling rules.
__device__ __forceinline__ float fast_tanh(float v) {
  float r;
  asm volatile("v_tanh_f32 %0, %1\n\tv_nop" : "=v"(r) : "v"(v));
  return r;
}

// ---------------------------------------------------------------------------
// 1a) Row sums -> 1/rowsum for fwd / rev normalized adjacency
// ---------------------------------------------------------------------------
__global__ void __launch_bounds__(256) prep_adj_inv(const float* __restrict__ adj,
                                                    float* __restrict__ invf,
                                                    float* __restrict__ invr) {
  int n   = blockIdx.x & (N_ - 1);
  int dir = blockIdx.x >> 9;
  __shared__ float red[256];
  float s = 0.f;
  for (int m = threadIdx.x; m < N_; m += 256) {
    float v = dir ? adj[(size_t)m * N_ + n] : adj[(size_t)n * N_ + m];
    if (m == n) v += 1.0f;
    s += v;
  }
  red[threadIdx.x] = s;
  __syncthreads();
  for (int st = 128; st > 0; st >>= 1) {
    if (threadIdx.x < st) red[threadIdx.x] += red[threadIdx.x + st];
    __syncthreads();
  }
  if (threadIdx.x == 0)
    (dir ? invr : invf)[n] = 1.0f / fmaxf(red[0], 1e-9f);
}

// ---------------------------------------------------------------------------
// 1b) Swizzle normalized adjacencies into WMMA A-fragment order:
//     adjsw[ntile(32)][kt(16)][lane(32)][16]
// ---------------------------------------------------------------------------
__global__ void __launch_bounds__(256) prep_adj_sw(const float* __restrict__ adj,
                                                   const float* __restrict__ invf,
                                                   const float* __restrict__ invr,
                                                   __bf16* __restrict__ adjswf,
                                                   __bf16* __restrict__ adjswr) {
  const int total = 32 * 16 * 32 * 16;  // 262144 per direction
  for (int idx = blockIdx.x * 256 + threadIdx.x; idx < 2 * total;
       idx += gridDim.x * 256) {
    int dir = idx >= total;
    int t = idx - dir * total;
    int hh = t & 15, lane = (t >> 4) & 31, kt = (t >> 9) & 15, ntile = t >> 13;
    int n = ntile * 16 + (lane & 15);
    int m = kt * 32 + kmapA(lane, hh);
    float v = dir ? adj[(size_t)m * N_ + n] : adj[(size_t)n * N_ + m];
    if (m == n) v += 1.0f;
    v *= (dir ? invr : invf)[n];
    (dir ? adjswr : adjswf)[t] = (__bf16)v;
  }
}

// ---------------------------------------------------------------------------
// 2) Weight packing (fragment order).  Mixprop algebra fold (P1=A@h, P2=A@P1):
//    out_dir = h@(W0+aW1+aW2) + P1@((1-a)W1+a(1-a)W2) + P2@((1-a)^2 W2)
// ---------------------------------------------------------------------------
__global__ void __launch_bounds__(256) prep_weights(
    const float* wf0, const float* wg0, const float* wf1, const float* wg1,
    const float* wf2, const float* wg2, const float* wf3, const float* wg3,
    const float* bf0, const float* bf1, const float* bf2, const float* bf3,
    const float* bg0, const float* bg1, const float* bg2, const float* bg3,
    const float* Wgf, const float* Wgr, const float* Ws,
    const float* bgf, const float* bgr,
    __bf16* Wcatsw,                                  // [14][8][32][16]
    __bf16* w0sw, __bf16* wssw,                      // [4][2][32][16] each
    __bf16* w1fsw, __bf16* w2fsw, __bf16* w1rsw, __bf16* w2rsw,
    float* bfcat, float* bgcat, float* bsum) {
  const int ks[4] = {2, 3, 6, 7};
  const float* wfs[4] = {wf0, wf1, wf2, wf3};
  const float* wgs[4] = {wg0, wg1, wg2, wg3};
  const float* bfs[4] = {bf0, bf1, bf2, bf3};
  const float* bgs[4] = {bg0, bg1, bg2, bg3};

  for (int t = threadIdx.x; t < 14 * 8 * 32 * 16; t += blockDim.x) {
    int hh = t & 15, lane = (t >> 4) & 31, nt = (t >> 9) & 7, kt = t >> 12;
    int k = kt * 32 + kmapB(lane, hh);
    int col = nt * 16 + (lane & 15);
    int d = k >> 6, cin = k & 63;
    int br = (col & 63) >> 4, o = col & 15;
    int kk = ks[br];
    float v = 0.f;
    if (d < kk) {
      const float* w = (col < 64) ? wfs[br] : wgs[br];
      v = w[((size_t)(kk - 1 - d) * 64 + cin) * 16 + o];
    }
    Wcatsw[t] = (__bf16)v;
  }
  const float a1 = (1.0f - ALPHA_);
  const float a12 = ALPHA_ * (1.0f - ALPHA_);
  const float a2 = (1.0f - ALPHA_) * (1.0f - ALPHA_);
  for (int idx = threadIdx.x; idx < 6 * 4096; idx += blockDim.x) {
    int mat = idx >> 12, t = idx & 4095;
    int hh = t & 15, lane = (t >> 4) & 31, kt = (t >> 9) & 1, ct = t >> 10;
    int k = kt * 32 + kmapB(lane, hh);
    int c = ct * 16 + (lane & 15);
    float v;
    float f1 = Wgf[(size_t)(64 + k) * 64 + c], f2 = Wgf[(size_t)(128 + k) * 64 + c];
    float r1 = Wgr[(size_t)(64 + k) * 64 + c], r2 = Wgr[(size_t)(128 + k) * 64 + c];
    switch (mat) {
      case 0: v = Wgf[(size_t)k * 64 + c] + Wgr[(size_t)k * 64 + c] +
                  ALPHA_ * (f1 + f2 + r1 + r2);           break;
      case 1: v = Ws[(size_t)k * 64 + c];                 break;
      case 2: v = a1 * f1 + a12 * f2;                     break;
      case 3: v = a2 * f2;                                break;
      case 4: v = a1 * r1 + a12 * r2;                     break;
      default: v = a2 * r2;                               break;
    }
    __bf16* dst[6] = {w0sw, wssw, w1fsw, w2fsw, w1rsw, w2rsw};
    dst[mat][t] = (__bf16)v;
  }
  for (int c = threadIdx.x; c < 64; c += blockDim.x) {
    bfcat[c] = bfs[c >> 4][c & 15];
    bgcat[c] = bgs[c >> 4][c & 15];
    bsum[c]  = bgf[c] + bgr[c];
  }
}

// ---------------------------------------------------------------------------
// 3) Gated temporal conv as GEMM (per wave 16 t-rows x 128 cols, K=448)
// ---------------------------------------------------------------------------
__global__ void __launch_bounds__(128) gated_conv(
    const float* __restrict__ x, const __bf16* __restrict__ Wcatsw,
    const float* __restrict__ bfcat, const float* __restrict__ bgcat,
    __bf16* __restrict__ h) {
  int bid   = blockIdx.x;                 // B * N * (T/64) = 16384
  int ttile = bid & 3;
  int n     = (bid >> 2) & (N_ - 1);
  int b     = bid >> 11;
  int t0    = ttile * 64;

  __shared__ __align__(16) __bf16 sX[70 * 64];
  int tid = threadIdx.x;
  for (int chunk = tid; chunk < 70 * 16; chunk += 128) {
    int e = chunk * 4;
    int row = e >> 6, c = e & 63;
    int t = t0 - 6 + row;
    __bf16* d = &sX[e];
    if (t >= 0) {
      float4 v = *(const float4*)&x[(((size_t)b * T_ + t) * N_ + n) * C_ + c];
      d[0] = (__bf16)v.x; d[1] = (__bf16)v.y; d[2] = (__bf16)v.z; d[3] = (__bf16)v.w;
    } else {
      d[0] = d[1] = d[2] = d[3] = (__bf16)0.f;
    }
  }
  __syncthreads();

  int lane = tid & 31, wave = tid >> 5;
  int toff = wave * 16;
  int mrow = lane & 15;
  int rbase = (lane & 16) ? 8 : 0;
  int choff = (lane & 16) ? 8 : 0;

  const v8f z = {0.f, 0.f, 0.f, 0.f, 0.f, 0.f, 0.f, 0.f};
  v8f acc[8];
#pragma unroll
  for (int nt = 0; nt < 8; ++nt) {
    v8f c8 = z;
    for (int kt = 0; kt < 14; ++kt) {
      Frag a, bf;
      int K0 = kt * 32 + choff;
      int d0 = K0 >> 6, ci0 = K0 & 63;
      int K1 = K0 + 16;
      int d1 = K1 >> 6, ci1 = K1 & 63;
      a.q[0] = *(const v4u*)&sX[(toff + mrow - d0 + 6) * 64 + ci0];
      a.q[1] = *(const v4u*)&sX[(toff + mrow - d1 + 6) * 64 + ci1];
      const __bf16* bb = Wcatsw + (((size_t)kt * 8 + nt) * 32 + lane) * 16;
      bf.q[0] = *(const v4u*)bb;
      bf.q[1] = *(const v4u*)(bb + 8);
      c8 = wmma_bf16(a, bf, c8);
    }
    acc[nt] = c8;
  }

#pragma unroll
  for (int ctf = 0; ctf < 4; ++ctf) {
    int ch = ctf * 16 + (lane & 15);
    float bfv = bfcat[ch], bgv = bgcat[ch];
#pragma unroll
    for (int r = 0; r < 8; ++r) {
      float f = acc[ctf][r] + bfv;
      float g = acc[ctf + 4][r] + bgv;
      float hv = fast_tanh(f) * (0.5f + 0.5f * fast_tanh(0.5f * g));
      int t = t0 + toff + r + rbase;
      h[(((size_t)b * T_ + t) * N_ + n) * C_ + ch] = (__bf16)hv;
    }
  }
}

// ---------------------------------------------------------------------------
// 4) h [row][c] -> hT [bt][c][n]
// ---------------------------------------------------------------------------
__global__ void __launch_bounds__(256) transpose_h(const __bf16* __restrict__ h,
                                                   __bf16* __restrict__ hT) {
  int bid = blockIdx.x;                   // 2048 bt * 8 ntile
  int nt = bid & 7, bt = bid >> 3;
  __shared__ __align__(16) __bf16 s[64 * 64];
  int tid = threadIdx.x;
  const v4u* p = (const v4u*)(h + ((size_t)bt * N_ + nt * 64) * C_);
  v4u* d = (v4u*)s;
  d[tid] = p[tid];
  d[tid + 256] = p[tid + 256];
  __syncthreads();
  int c = tid >> 2, n0 = (tid & 3) * 16;
  union { v4u q; __bf16 e[8]; } p0, p1;
#pragma unroll
  for (int j = 0; j < 8; ++j) {
    p0.e[j] = s[(n0 + j) * 64 + c];
    p1.e[j] = s[(n0 + 8 + j) * 64 + c];
  }
  v4u* dst = (v4u*)(hT + (size_t)bt * (N_ * C_) + (size_t)c * N_ + nt * 64 + n0);
  dst[0] = p0.q;
  dst[1] = p1.q;
}

// ---------------------------------------------------------------------------
// 5) skip = h @ Ws + bs
// ---------------------------------------------------------------------------
__global__ void __launch_bounds__(256) skip_kernel(
    const __bf16* __restrict__ h, const __bf16* __restrict__ wssw,
    const float* __restrict__ bs, float* __restrict__ skip) {
  size_t wglob = ((size_t)blockIdx.x * 256 + threadIdx.x) >> 5;
  size_t r0 = wglob * 16;
  int lane = threadIdx.x & 31;
  int mrow = lane & 15;
  int rbase = (lane & 16) ? 8 : 0;
  int choff = (lane & 16) ? 8 : 0;

  const v8f z = {0.f, 0.f, 0.f, 0.f, 0.f, 0.f, 0.f, 0.f};
  v8f acs[4] = {z, z, z, z};
#pragma unroll
  for (int kt = 0; kt < 2; ++kt) {
    Frag a;
    const __bf16* ap = h + (r0 + mrow) * C_ + kt * 32 + choff;
    a.q[0] = *(const v4u*)ap;
    a.q[1] = *(const v4u*)(ap + 16);
#pragma unroll
    for (int ct = 0; ct < 4; ++ct) {
      Frag b;
      const __bf16* bp = wssw + (((size_t)ct * 2 + kt) * 32 + lane) * 16;
      b.q[0] = *(const v4u*)bp;
      b.q[1] = *(const v4u*)(bp + 8);
      acs[ct] = wmma_bf16(a, b, acs[ct]);
    }
  }
#pragma unroll
  for (int ct = 0; ct < 4; ++ct) {
    int c = ct * 16 + (lane & 15);
    float bsv = bs[c];
#pragma unroll
    for (int r = 0; r < 8; ++r)
      skip[(r0 + r + rbase) * C_ + c] = acs[ct][r] + bsv;
  }
}

// ---------------------------------------------------------------------------
// 6) Fully fused graph output per (b,t): one 256-thread block owns the whole
//    512-node slab.  TDM stages h^T; P1 lives only in LDS; all projection
//    terms accumulate in registers; `out` is written exactly once.
//    Dynamic LDS: sHT 32KB | sP1T 64KB | sO 16KB  = 112KB.
// ---------------------------------------------------------------------------
__global__ void __launch_bounds__(256) graph_out(
    const float* __restrict__ x, const __bf16* __restrict__ h,
    const __bf16* __restrict__ hT,
    const __bf16* __restrict__ adjswf, const __bf16* __restrict__ adjswr,
    const __bf16* __restrict__ w0sw,
    const __bf16* __restrict__ w1fsw, const __bf16* __restrict__ w2fsw,
    const __bf16* __restrict__ w1rsw, const __bf16* __restrict__ w2rsw,
    const float* __restrict__ bsum, float* __restrict__ out) {
  int bt = blockIdx.x;                   // 2048 blocks, 8 waves each
  extern __shared__ __align__(16) char smem[];
  __bf16* sHT  = (__bf16*)smem;                    // [64][256] phase slab
  __bf16* sP1T = (__bf16*)(smem + 32768);          // [64][512] hop-1 result
  __bf16* sO   = (__bf16*)(smem + 32768 + 65536);  // [8][16*64] reshape tiles

  int tid = threadIdx.x, lane = tid & 31, wave = tid >> 5;
  int mrow = lane & 15;
  int rbase = (lane & 16) ? 8 : 0;
  int choff = (lane & 16) ? 8 : 0;
  int boff = (lane & 16) ? 16 : 0;
  __bf16* sOw = sO + wave * 1024;

  const v8f z = {0.f, 0.f, 0.f, 0.f, 0.f, 0.f, 0.f, 0.f};
  v8f oacc[4][4];                        // running out tiles (rows n, cols c)
  v8f acc[4][4];                         // per-hop A@H accumulators

  // --- oacc = h @ w0'  (A-frags from row-major h; fully coherent loads)
#pragma unroll
  for (int nc = 0; nc < 4; ++nc) {
    size_t r0 = (size_t)bt * N_ + (nc * 8 + wave) * 16;
#pragma unroll
    for (int ct = 0; ct < 4; ++ct) oacc[nc][ct] = z;
#pragma unroll
    for (int kt = 0; kt < 2; ++kt) {
      Frag a;
      const __bf16* ap = h + (r0 + mrow) * C_ + kt * 32 + choff;
      a.q[0] = *(const v4u*)ap;
      a.q[1] = *(const v4u*)(ap + 16);
#pragma unroll
      for (int ct = 0; ct < 4; ++ct) {
        Frag b;
        const __bf16* bp = w0sw + (((size_t)ct * 2 + kt) * 32 + lane) * 16;
        b.q[0] = *(const v4u*)bp;
        b.q[1] = *(const v4u*)(bp + 8);
        oacc[nc][ct] = wmma_bf16(a, b, oacc[nc][ct]);
      }
    }
  }

  for (int dir = 0; dir < 2; ++dir) {
    const __bf16* adjsw = dir ? adjswr : adjswf;
    const __bf16* w1 = dir ? w1rsw : w1fsw;
    const __bf16* w2 = dir ? w2rsw : w2fsw;

    // ---- hop 1: acc = A @ h   (B staged by Tensor Data Mover, 2 phases)
#pragma unroll
    for (int nc = 0; nc < 4; ++nc)
#pragma unroll
      for (int ct = 0; ct < 4; ++ct) acc[nc][ct] = z;

    for (int ph = 0; ph < 2; ++ph) {
      __syncthreads();
      if (tid < 32) {
        unsigned lds = (unsigned)(unsigned long long)(void*)sHT;
        unsigned long long ga =
            (unsigned long long)(hT + (size_t)bt * (N_ * C_) + ph * 256);
        u32x4 g0 = {1u, lds, (unsigned)ga,
                    (unsigned)((ga >> 32) & 0x1FFFFFFu) | 0x80000000u};
        i32x8 g1 = {(int)(1u << 16),   (int)(512u << 16), (int)(64u << 16),
                    (int)(256u << 16), 64, 512, 0, 0};
        i32x4 gz = {0, 0, 0, 0};
#if defined(__clang_major__) && __clang_major__ >= 23
        i32x8 gz8 = {0, 0, 0, 0, 0, 0, 0, 0};
        __builtin_amdgcn_tensor_load_to_lds(g0, g1, gz, gz, gz8, 0);
#else
        __builtin_amdgcn_tensor_load_to_lds(g0, g1, gz, gz, 0);
#endif
        __builtin_amdgcn_s_wait_tensorcnt(0);
      }
      __syncthreads();
      for (int kt = 0; kt < 8; ++kt) {
#pragma unroll
        for (int nc = 0; nc < 4; ++nc) {
          int ntile = nc * 8 + wave;
          Frag a;
          const __bf16* ap =
              adjsw + (((size_t)ntile * 16 + (ph * 8 + kt)) * 32 + lane) * 16;
          a.q[0] = *(const v4u*)ap;
          a.q[1] = *(const v4u*)(ap + 8);
#pragma unroll
          for (int ct = 0; ct < 4; ++ct) {
            Frag b;
            const __bf16* bp =
                &sHT[(ct * 16 + (lane & 15)) * 256 + kt * 32 + boff];
            b.q[0] = *(const v4u*)bp;
            b.q[1] = *(const v4u*)(bp + 8);
            acc[nc][ct] = wmma_bf16(a, b, acc[nc][ct]);
          }
        }
      }
    }

    // ---- P1 -> sP1T (vector ds stores) ; oacc += P1 @ w1 (via sO reshape)
    for (int nc = 0; nc < 4; ++nc) {
      int nbase = (nc * 8 + wave) * 16;
#pragma unroll
      for (int ct = 0; ct < 4; ++ct) {
        int c = ct * 16 + (lane & 15);
        union { v4u q; __bf16 e[8]; } pk;
#pragma unroll
        for (int r = 0; r < 8; ++r) {
          __bf16 pv = (__bf16)acc[nc][ct][r];
          pk.e[r] = pv;
          sOw[(r + rbase) * 64 + c] = pv;
        }
        *(v4u*)&sP1T[(size_t)c * N_ + nbase + rbase] = pk.q;
      }
#pragma unroll
      for (int ct2 = 0; ct2 < 4; ++ct2) {
#pragma unroll
        for (int kt2 = 0; kt2 < 2; ++kt2) {
          Frag a2, b2;
          const __bf16* ap = &sOw[mrow * 64 + kt2 * 32 + choff];
          a2.q[0] = *(const v4u*)ap;
          a2.q[1] = *(const v4u*)(ap + 16);
          const __bf16* bp = w1 + (((size_t)ct2 * 2 + kt2) * 32 + lane) * 16;
          b2.q[0] = *(const v4u*)bp;
          b2.q[1] = *(const v4u*)(bp + 8);
          oacc[nc][ct2] = wmma_bf16(a2, b2, oacc[nc][ct2]);
        }
      }
    }
    __syncthreads();                      // sP1T complete for all waves

    // ---- hop 2: acc = A @ P1  (B straight from LDS sP1T)
#pragma unroll
    for (int nc = 0; nc < 4; ++nc)
#pragma unroll
      for (int ct = 0; ct < 4; ++ct) acc[nc][ct] = z;
    for (int kt = 0; kt < 16; ++kt) {
#pragma unroll
      for (int nc = 0; nc < 4; ++nc) {
        int ntile = nc * 8 + wave;
        Frag a;
        const __bf16* ap =
            adjsw + (((size_t)ntile * 16 + kt) * 32 + lane) * 16;
        a.q[0] = *(const v4u*)ap;
        a.q[1] = *(const v4u*)(ap + 8);
#pragma unroll
        for (int ct = 0; ct < 4; ++ct) {
          Frag b;
          const __bf16* bp =
              &sP1T[(size_t)(ct * 16 + (lane & 15)) * N_ + kt * 32 + boff];
          b.q[0] = *(const v4u*)bp;
          b.q[1] = *(const v4u*)(bp + 8);
          acc[nc][ct] = wmma_bf16(a, b, acc[nc][ct]);
        }
      }
    }
    // ---- oacc += P2 @ w2
    for (int nc = 0; nc < 4; ++nc) {
#pragma unroll
      for (int ct = 0; ct < 4; ++ct) {
        int c = ct * 16 + (lane & 15);
#pragma unroll
        for (int r = 0; r < 8; ++r)
          sOw[(r + rbase) * 64 + c] = (__bf16)acc[nc][ct][r];
      }
#pragma unroll
      for (int ct2 = 0; ct2 < 4; ++ct2) {
#pragma unroll
        for (int kt2 = 0; kt2 < 2; ++kt2) {
          Frag a2, b2;
          const __bf16* ap = &sOw[mrow * 64 + kt2 * 32 + choff];
          a2.q[0] = *(const v4u*)ap;
          a2.q[1] = *(const v4u*)(ap + 16);
          const __bf16* bp = w2 + (((size_t)ct2 * 2 + kt2) * 32 + lane) * 16;
          b2.q[0] = *(const v4u*)bp;
          b2.q[1] = *(const v4u*)(bp + 8);
          oacc[nc][ct2] = wmma_bf16(a2, b2, oacc[nc][ct2]);
        }
      }
    }
    // next dir's phase barriers protect sHT/sP1T reuse
  }

  // ---- single write: out = oacc + x + bsum
#pragma unroll
  for (int nc = 0; nc < 4; ++nc) {
    size_t rowbase = (size_t)bt * N_ + (nc * 8 + wave) * 16;
#pragma unroll
    for (int ct = 0; ct < 4; ++ct) {
      int c = ct * 16 + (lane & 15);
      float bo = bsum[c];
#pragma unroll
      for (int r = 0; r < 8; ++r) {
        size_t o = (rowbase + r + rbase) * C_ + c;
        out[o] = oacc[nc][ct][r] + x[o] + bo;
      }
    }
  }
}

// ---------------------------------------------------------------------------
// Host-side orchestration
// ---------------------------------------------------------------------------
static inline size_t align256(size_t v) { return (v + 255) & ~(size_t)255; }

extern "C" void kernel_launch(void* const* d_in, const int* in_sizes, int n_in,
                              void* d_out, int out_size, void* d_ws, size_t ws_size,
                              hipStream_t stream) {
  (void)in_sizes; (void)n_in; (void)out_size; (void)ws_size;
  const float* x   = (const float*)d_in[0];
  const float* adj = (const float*)d_in[1];
  const float* wf[4], *bf[4], *wg[4], *bg[4];
  for (int i = 0; i < 4; ++i) {
    wf[i] = (const float*)d_in[2 + i * 4 + 0];
    bf[i] = (const float*)d_in[2 + i * 4 + 1];
    wg[i] = (const float*)d_in[2 + i * 4 + 2];
    bg[i] = (const float*)d_in[2 + i * 4 + 3];
  }
  const float* Wgf = (const float*)d_in[18];
  const float* bgf = (const float*)d_in[19];
  const float* Wgr = (const float*)d_in[20];
  const float* bgr = (const float*)d_in[21];
  const float* Ws  = (const float*)d_in[22];
  const float* bs  = (const float*)d_in[23];

  float* out  = (float*)d_out;
  float* skip = out + ROWS * C_;

  char* ws = (char*)d_ws;
  size_t off = 0;
  const size_t H_BYTES = ROWS * C_ * sizeof(__bf16);        // 128 MB
  __bf16* hbuf   = (__bf16*)(ws + off); off = align256(off + H_BYTES);
  __bf16* hTbuf  = (__bf16*)(ws + off); off = align256(off + H_BYTES);
  __bf16* adjswf = (__bf16*)(ws + off); off = align256(off + (size_t)N_ * N_ * 2);
  __bf16* adjswr = (__bf16*)(ws + off); off = align256(off + (size_t)N_ * N_ * 2);
  __bf16* Wcatsw = (__bf16*)(ws + off); off = align256(off + 448 * 128 * 2);
  __bf16* w0sw   = (__bf16*)(ws + off); off = align256(off + 4096 * 2);
  __bf16* wssw   = (__bf16*)(ws + off); off = align256(off + 4096 * 2);
  __bf16* w1fsw  = (__bf16*)(ws + off); off = align256(off + 4096 * 2);
  __bf16* w2fsw  = (__bf16*)(ws + off); off = align256(off + 4096 * 2);
  __bf16* w1rsw  = (__bf16*)(ws + off); off = align256(off + 4096 * 2);
  __bf16* w2rsw  = (__bf16*)(ws + off); off = align256(off + 4096 * 2);
  float* invf    = (float*)(ws + off);  off = align256(off + 512 * 4);
  float* invr    = (float*)(ws + off);  off = align256(off + 512 * 4);
  float* bfcat   = (float*)(ws + off);  off = align256(off + 64 * 4);
  float* bgcat   = (float*)(ws + off);  off = align256(off + 64 * 4);
  float* bsum    = (float*)(ws + off);  off = align256(off + 64 * 4);

  prep_adj_inv<<<1024, 256, 0, stream>>>(adj, invf, invr);
  prep_adj_sw<<<512, 256, 0, stream>>>(adj, invf, invr, adjswf, adjswr);
  prep_weights<<<1, 256, 0, stream>>>(
      wf[0], wg[0], wf[1], wg[1], wf[2], wg[2], wf[3], wg[3],
      bf[0], bf[1], bf[2], bf[3], bg[0], bg[1], bg[2], bg[3],
      Wgf, Wgr, Ws, bgf, bgr,
      Wcatsw, w0sw, wssw, w1fsw, w2fsw, w1rsw, w2rsw, bfcat, bgcat, bsum);

  gated_conv<<<B_ * N_ * (T_ / 64), 128, 0, stream>>>(x, Wcatsw, bfcat, bgcat, hbuf);
  transpose_h<<<B_ * T_ * 8, 256, 0, stream>>>(hbuf, hTbuf);
  skip_kernel<<<(int)(ROWS / 16 / 8), 256, 0, stream>>>(hbuf, wssw, bs, skip);

  const size_t GRAPH_LDS = 32768 + 65536 + 16384;   // 112 KB dynamic LDS
  graph_out<<<B_ * T_, 256, GRAPH_LDS, stream>>>(
      x, hbuf, hTbuf, adjswf, adjswr, w0sw,
      w1fsw, w2fsw, w1rsw, w2rsw, bsum, out);
}
